// FcGagaLayer_71012989272234
// MI455X (gfx1250) — matrix operand order repaired and checked by reading, not compiled, not run on hardware
//
#include <hip/hip_runtime.h>
#include <hip/hip_bf16.h>
#include <math.h>

#define Bc      32
#define Nn      207
#define Tt      12
#define Cc      2
#define HORc    12
#define DIDc    64
#define Hc      128
#define NBc     3
#define CONCATc 2560            // 12 + 207*12 + 64
#define Rrows   (Bc*Nn*Cc)      // 13248 = 414*32

typedef __attribute__((ext_vector_type(16))) __bf16 bf16x16;
typedef __attribute__((ext_vector_type(8)))  float  f32x8;

__device__ __forceinline__ unsigned short f2bf(float f) {
    unsigned int u = __float_as_uint(f);
    if ((u & 0x7fffffffu) > 0x7f800000u) return (unsigned short)0x7fc0u; // NaN
    unsigned int r = u + 0x7fffu + ((u >> 16) & 1u);                     // RNE
    return (unsigned short)(r >> 16);
}

__device__ __forceinline__ float dnn(float a, float b) {
    float r = a / (b + 1e-8f);
    return __builtin_isfinite(r) ? r : 0.0f;
}

// ---------------- weight transpose + bf16 convert: src (K x Nsrc) -> dst (Ndst x K)
__global__ void conv_tr_kernel(const float* __restrict__ src, unsigned short* __restrict__ dst,
                               int K, int Nsrc, int Ndst) {
    int e = blockIdx.x * blockDim.x + threadIdx.x;
    if (e >= Ndst * K) return;
    int k = e % K, n = e / K;
    dst[(size_t)n * K + k] = (n < Nsrc) ? f2bf(src[(size_t)k * Nsrc + n]) : (unsigned short)0;
}

// ---------------- dp = exp(10 * emb @ emb^T)
__global__ void dp_kernel(const float* __restrict__ emb, float* __restrict__ dp) {
    int e = blockIdx.x * blockDim.x + threadIdx.x;
    if (e >= Nn * Nn) return;
    int n2 = e % Nn, n1 = e / Nn;
    float a = 0.f;
    for (int d = 0; d < DIDc; ++d) a += emb[n1 * DIDc + d] * emb[n2 * DIDc + d];
    dp[e] = expf(10.0f * a);
}

// ---------------- time gates: one block (128 thr) per (b,n)
__global__ void timegate_kernel(const float* __restrict__ tod, const float* __restrict__ emb,
                                const float* __restrict__ w1, const float* __restrict__ b1,
                                const float* __restrict__ w2, const float* __restrict__ b2,
                                const float* __restrict__ w3, const float* __restrict__ b3,
                                float* __restrict__ tgf, float* __restrict__ tgb) {
    int bn = blockIdx.x;
    int n  = bn % Nn;
    int tid = threadIdx.x;
    __shared__ float in[DIDc + Tt];
    __shared__ float hid[Hc];
    if (tid < DIDc)            in[tid] = emb[n * DIDc + tid];
    else if (tid < DIDc + Tt)  in[tid] = tod[bn * Tt + (tid - DIDc)];
    __syncthreads();
    float acc = b1[tid];
    for (int k = 0; k < DIDc + Tt; ++k) acc += in[k] * w1[k * Hc + tid];
    hid[tid] = fmaxf(acc, 0.f);
    __syncthreads();
    if (tid < HORc) {
        float a = b2[tid];
        for (int h = 0; h < Hc; ++h) a += hid[h] * w2[h * HORc + tid];
        tgf[bn * HORc + tid] = a;
    } else if (tid >= 32 && tid < 32 + Tt) {
        int t = tid - 32;
        float a = b3[t];
        for (int h = 0; h < Hc; ++h) a += hid[h] * w3[h * Tt + t];
        tgb[bn * Tt + t] = a;
    }
}

// ---------------- h_in = history/(1+tgb); level = max_t h_in
__global__ void hin_level_kernel(const float* __restrict__ hist, const float* __restrict__ tgb,
                                 float* __restrict__ h_in, float* __restrict__ level) {
    int bn = blockIdx.x * blockDim.x + threadIdx.x;
    if (bn >= Bc * Nn) return;
    float mx0 = -INFINITY, mx1 = -INFINITY;
    for (int t = 0; t < Tt; ++t) {
        float g  = 1.0f + tgb[bn * Tt + t];
        float h0 = hist[(bn * Tt + t) * Cc + 0] / g;
        float h1 = hist[(bn * Tt + t) * Cc + 1] / g;
        h_in[(bn * Tt + t) * Cc + 0] = h0;
        h_in[(bn * Tt + t) * Cc + 1] = h1;
        mx0 = fmaxf(mx0, h0); mx1 = fmaxf(mx1, h1);
    }
    level[bn * Cc + 0] = mx0;
    level[bn * Cc + 1] = mx1;
}

// ---------------- build X = (B,N,C,CONCAT) fp32 + bf16 mirror
__global__ void build_x_kernel(const float* __restrict__ h_in, const float* __restrict__ level,
                               const float* __restrict__ dp, const float* __restrict__ emb,
                               float* __restrict__ Xf, unsigned short* __restrict__ Xb) {
    long long e = (long long)blockIdx.x * blockDim.x + threadIdx.x;
    if (e >= (long long)Bc * Nn * CONCATc) return;
    int k  = (int)(e % CONCATc);
    int bn = (int)(e / CONCATc);
    int n  = bn % Nn, b = bn / Nn;
    float l0 = level[bn * Cc + 0], l1 = level[bn * Cc + 1];
    float v0, v1;
    if (k < Tt) {
        v0 = dnn(h_in[(bn * Tt + k) * Cc + 0], l0);
        v1 = dnn(h_in[(bn * Tt + k) * Cc + 1], l1);
    } else if (k < Tt + Nn * Tt) {
        int q = k - Tt;
        int n2 = q / Tt, t = q - n2 * Tt;
        float w = dp[n * Nn + n2];
        int src = ((b * Nn + n2) * Tt + t) * Cc;
        v0 = fmaxf(dnn(h_in[src + 0] * w - l0, l0), 0.f);
        v1 = fmaxf(dnn(h_in[src + 1] * w - l1, l1), 0.f);
    } else {
        float ne = emb[n * DIDc + (k - (Tt + Nn * Tt))];
        v0 = ne; v1 = ne;
    }
    size_t o0 = ((size_t)bn * Cc + 0) * CONCATc + k;
    size_t o1 = ((size_t)bn * Cc + 1) * CONCATc + k;
    Xf[o0] = v0; Xf[o1] = v1;
    Xb[o0] = f2bf(v0); Xb[o1] = f2bf(v1);
}

union FragBF { bf16x16 v; uint4 q[2]; };

#define WMMA_BF16(a, b, c) \
    __builtin_amdgcn_wmma_f32_16x16x32_bf16(false, (a), false, (b), (short)0, (c), false, false)

// ---------------- 2x2 register-blocked WMMA GEMM: one wave per 32x32 output macro-tile.
// A: (Rrows x K) bf16 row-major.  Wt: (Nout x K) bf16 row-major (pre-transposed).
// mode 0: out_bf = bf16(relu(A@W + bias))
// mode 1: v = relu(Xf - (A@W + bias)); Xf = v; out_bf = bf16(v)
// Requires Nout % 32 == 0.
__global__ void wmma_gemm2x2_kernel(const unsigned short* __restrict__ A, int K,
                                    const unsigned short* __restrict__ Wt,
                                    const float* __restrict__ bias, int Nout,
                                    int mode, float* __restrict__ xf,
                                    unsigned short* __restrict__ outb) {
    int wid    = blockIdx.x * (blockDim.x >> 5) + (threadIdx.x >> 5);
    int lane   = threadIdx.x & 31;
    int ntiles = Nout >> 5;
    int total  = (Rrows / 32) * ntiles;
    if (wid >= total) return;                 // wave-uniform exit
    int m0   = (wid / ntiles) * 32;
    int n0   = (wid % ntiles) * 32;
    int half = lane >> 4;                     // 0: lanes 0-15, 1: lanes 16-31
    int l16  = lane & 15;

    // A fragment per ISA 16-bit 16x32 layout: half0 -> K 0-7 & 16-23, half1 -> K 8-15 & 24-31
    const unsigned short* arow0 = A + (size_t)(m0 + l16) * K + (half ? 8 : 0);
    const unsigned short* arow1 = arow0 + (size_t)16 * K;
    // B fragment: same K split as A: half0 -> K 0-7 & 16-23, half1 -> K 8-15 & 24-31
    const unsigned short* brow0 = Wt + (size_t)(n0 + l16) * K + (half ? 8 : 0);
    const unsigned short* brow1 = brow0 + (size_t)16 * K;

    f32x8 c00 = {}, c01 = {}, c10 = {}, c11 = {};
    for (int k0 = 0; k0 < K; k0 += 32) {
        FragBF a0, a1, b0, b1;
        a0.q[0] = *(const uint4*)(arow0 + k0);
        a0.q[1] = *(const uint4*)(arow0 + k0 + 16);
        a1.q[0] = *(const uint4*)(arow1 + k0);
        a1.q[1] = *(const uint4*)(arow1 + k0 + 16);
        b0.q[0] = *(const uint4*)(brow0 + k0);
        b0.q[1] = *(const uint4*)(brow0 + k0 + 16);
        b1.q[0] = *(const uint4*)(brow1 + k0);
        b1.q[1] = *(const uint4*)(brow1 + k0 + 16);
        c00 = WMMA_BF16(a0.v, b0.v, c00);
        c01 = WMMA_BF16(a0.v, b1.v, c01);
        c10 = WMMA_BF16(a1.v, b0.v, c10);
        c11 = WMMA_BF16(a1.v, b1.v, c11);
    }

    f32x8 accs[4] = { c00, c01, c10, c11 };
    for (int si = 0; si < 2; ++si) {
        for (int sj = 0; sj < 2; ++sj) {
            f32x8 c = accs[si * 2 + sj];
            int n = n0 + sj * 16 + l16;
            float bv = bias[n];
            for (int r = 0; r < 8; ++r) {
                int m = m0 + si * 16 + r + half * 8;
                size_t o = (size_t)m * Nout + n;
                if (mode == 1) {
                    float v = fmaxf(xf[o] - (c[r] + bv), 0.f);
                    xf[o] = v;
                    outb[o] = f2bf(v);
                } else {
                    float v = fmaxf(c[r] + bv, 0.f);
                    outb[o] = f2bf(v);
                }
            }
        }
    }
}

// ---------------- single-tile WMMA GEMM for the forecast head (Nout == 16, cols < HOR valid)
__global__ void wmma_gemm_fore_kernel(const unsigned short* __restrict__ A, int K,
                                      const unsigned short* __restrict__ Wt,
                                      const float* __restrict__ bias,
                                      int accflag, float* __restrict__ facc) {
    int wid  = blockIdx.x * (blockDim.x >> 5) + (threadIdx.x >> 5);
    int lane = threadIdx.x & 31;
    int total = Rrows / 16;
    if (wid >= total) return;                 // wave-uniform exit
    int m0   = wid * 16;
    int half = lane >> 4;
    int l16  = lane & 15;

    const unsigned short* arow = A  + (size_t)(m0 + l16) * K + (half ? 8 : 0);
    const unsigned short* brow = Wt + (size_t)l16 * K + (half ? 8 : 0);

    f32x8 c = {};
    for (int k0 = 0; k0 < K; k0 += 32) {
        FragBF a, b;
        a.q[0] = *(const uint4*)(arow + k0);
        a.q[1] = *(const uint4*)(arow + k0 + 16);
        b.q[0] = *(const uint4*)(brow + k0);
        b.q[1] = *(const uint4*)(brow + k0 + 16);
        c = WMMA_BF16(a.v, b.v, c);
    }

    int n = l16;
    if (n < HORc) {
        float bv = bias[n];
        for (int r = 0; r < 8; ++r) {
            int m = m0 + r + half * 8;
            float v = c[r] + bv;
            size_t o = (size_t)m * 16 + n;
            facc[o] = accflag ? (facc[o] + v) : v;
        }
    }
}

// ---------------- final writes
__global__ void write_backcast_kernel(const float* __restrict__ xf, float* __restrict__ out) {
    long long e = (long long)blockIdx.x * blockDim.x + threadIdx.x;
    if (e >= (long long)Bc * Nn * CONCATc) return;
    int k  = (int)(e % CONCATc);
    long long bn = e / CONCATc;
    float v0 = xf[((size_t)bn * Cc + 0) * CONCATc + k];
    float v1 = xf[((size_t)bn * Cc + 1) * CONCATc + k];
    out[((size_t)bn * CONCATc + k) * Cc + 0] = v0;
    out[((size_t)bn * CONCATc + k) * Cc + 1] = v1;
}

__global__ void write_forecast_kernel(const float* __restrict__ facc, const float* __restrict__ level,
                                      const float* __restrict__ tgf, float* __restrict__ out) {
    int e = blockIdx.x * blockDim.x + threadIdx.x;
    if (e >= Bc * Nn * HORc * Cc) return;
    int c  = e & 1;
    int t  = (e >> 1) % HORc;
    int bn = e / (HORc * Cc);
    float v = facc[((size_t)bn * Cc + c) * 16 + t];
    v *= level[bn * Cc + c];
    v *= (1.0f + tgf[bn * HORc + t]);
    out[e] = v;
}

static inline int cdiv(long long a, long long b) { return (int)((a + b - 1) / b); }

extern "C" void kernel_launch(void* const* d_in, const int* in_sizes, int n_in,
                              void* d_out, int out_size, void* d_ws, size_t ws_size,
                              hipStream_t stream) {
    const float* history  = (const float*)d_in[0];
    const float* tod      = (const float*)d_in[1];
    const float* node_emb = (const float*)d_in[2];
    const float* tg1_w = (const float*)d_in[3];
    const float* tg1_b = (const float*)d_in[4];
    const float* tg2_w = (const float*)d_in[5];
    const float* tg2_b = (const float*)d_in[6];
    const float* tg3_w = (const float*)d_in[7];
    const float* tg3_b = (const float*)d_in[8];
    const float* fc0_w = (const float*)d_in[9];
    const float* fc0_b = (const float*)d_in[10];
    const float* fc_w  = (const float*)d_in[11];
    const float* fc_b  = (const float*)d_in[12];
    const float* fore_w = (const float*)d_in[13];
    const float* fore_b = (const float*)d_in[14];
    const float* back_w = (const float*)d_in[15];
    const float* back_b = (const float*)d_in[16];

    // ---- workspace carve-up (256B aligned)
    char* base = (char*)d_ws;
    size_t off = 0;
    auto carve = [&](size_t bytes) -> char* {
        off = (off + 255) & ~(size_t)255;
        char* p = base + off;
        off += bytes;
        return p;
    };
    float* dp      = (float*)carve((size_t)Nn * Nn * 4);
    float* tgf     = (float*)carve((size_t)Bc * Nn * HORc * 4);
    float* tgb     = (float*)carve((size_t)Bc * Nn * Tt * 4);
    float* level   = (float*)carve((size_t)Bc * Nn * Cc * 4);
    float* h_in    = (float*)carve((size_t)Bc * Nn * Tt * Cc * 4);
    float* Xf      = (float*)carve((size_t)Rrows * CONCATc * 4);
    unsigned short* Xb  = (unsigned short*)carve((size_t)Rrows * CONCATc * 2);
    unsigned short* hA  = (unsigned short*)carve((size_t)Rrows * Hc * 2);
    unsigned short* hB  = (unsigned short*)carve((size_t)Rrows * Hc * 2);
    float* facc    = (float*)carve((size_t)Rrows * 16 * 4);
    unsigned short* fc0_wt  = (unsigned short*)carve((size_t)NBc * Hc * CONCATc * 2);
    unsigned short* fc_wt   = (unsigned short*)carve((size_t)NBc * 2 * Hc * Hc * 2);
    unsigned short* back_wt = (unsigned short*)carve((size_t)NBc * CONCATc * Hc * 2);
    unsigned short* fore_wt = (unsigned short*)carve((size_t)NBc * 16 * Hc * 2);
    (void)ws_size; (void)in_sizes; (void)n_in; (void)out_size;

    // ---- weight prep (transpose -> bf16)
    for (int i = 0; i < NBc; ++i) {
        conv_tr_kernel<<<cdiv((long long)Hc * CONCATc, 256), 256, 0, stream>>>(
            fc0_w + (size_t)i * CONCATc * Hc, fc0_wt + (size_t)i * Hc * CONCATc,
            CONCATc, Hc, Hc);
        for (int j = 0; j < 2; ++j)
            conv_tr_kernel<<<cdiv((long long)Hc * Hc, 256), 256, 0, stream>>>(
                fc_w + (size_t)(i * 2 + j) * Hc * Hc, fc_wt + (size_t)(i * 2 + j) * Hc * Hc,
                Hc, Hc, Hc);
        conv_tr_kernel<<<cdiv((long long)CONCATc * Hc, 256), 256, 0, stream>>>(
            back_w + (size_t)i * Hc * CONCATc, back_wt + (size_t)i * CONCATc * Hc,
            Hc, CONCATc, CONCATc);
        conv_tr_kernel<<<cdiv((long long)16 * Hc, 256), 256, 0, stream>>>(
            fore_w + (size_t)i * Hc * HORc, fore_wt + (size_t)i * 16 * Hc,
            Hc, HORc, 16);
    }

    // ---- small prep kernels
    dp_kernel<<<cdiv((long long)Nn * Nn, 256), 256, 0, stream>>>(node_emb, dp);
    timegate_kernel<<<Bc * Nn, Hc, 0, stream>>>(tod, node_emb, tg1_w, tg1_b,
                                                tg2_w, tg2_b, tg3_w, tg3_b, tgf, tgb);
    hin_level_kernel<<<cdiv(Bc * Nn, 256), 256, 0, stream>>>(history, tgb, h_in, level);
    build_x_kernel<<<cdiv((long long)Bc * Nn * CONCATc, 256), 256, 0, stream>>>(
        h_in, level, dp, node_emb, Xf, Xb);

    // ---- N-BEATS blocks (2x2-blocked WMMA GEMMs, 8 waves / 256-thread block)
    const int MT32 = Rrows / 32; // 414
    for (int i = 0; i < NBc; ++i) {
        // fc0: X(13248x2560) @ W(2560x128) -> relu -> hA
        wmma_gemm2x2_kernel<<<cdiv((long long)MT32 * (Hc / 32), 8), 256, 0, stream>>>(
            Xb, CONCATc, fc0_wt + (size_t)i * Hc * CONCATc, fc0_b + (size_t)i * Hc,
            Hc, 0, nullptr, hA);
        // fc layer 0: hA @ W -> relu -> hB
        wmma_gemm2x2_kernel<<<cdiv((long long)MT32 * (Hc / 32), 8), 256, 0, stream>>>(
            hA, Hc, fc_wt + (size_t)(i * 2 + 0) * Hc * Hc, fc_b + (size_t)(i * 2 + 0) * Hc,
            Hc, 0, nullptr, hB);
        // fc layer 1: hB @ W -> relu -> hA
        wmma_gemm2x2_kernel<<<cdiv((long long)MT32 * (Hc / 32), 8), 256, 0, stream>>>(
            hB, Hc, fc_wt + (size_t)(i * 2 + 1) * Hc * Hc, fc_b + (size_t)(i * 2 + 1) * Hc,
            Hc, 0, nullptr, hA);
        // forecast: hA @ fore_w -> facc (accumulate across blocks)
        wmma_gemm_fore_kernel<<<cdiv((long long)(Rrows / 16), 8), 256, 0, stream>>>(
            hA, Hc, fore_wt + (size_t)i * 16 * Hc, fore_b + (size_t)i * HORc,
            (i > 0) ? 1 : 0, facc);
        // backcast: X = relu(X - (hA @ back_w + b)); refresh bf16 mirror
        wmma_gemm2x2_kernel<<<cdiv((long long)MT32 * (CONCATc / 32), 8), 256, 0, stream>>>(
            hA, Hc, back_wt + (size_t)i * CONCATc * Hc, back_b + (size_t)i * CONCATc,
            CONCATc, 1, Xf, Xb);
    }

    // ---- outputs: backcast (B,N,CONCAT,C) then forecast (B,N,HOR,C)
    float* out_back = (float*)d_out;
    float* out_fore = out_back + (size_t)Bc * Nn * CONCATc * Cc;
    write_backcast_kernel<<<cdiv((long long)Bc * Nn * CONCATc, 256), 256, 0, stream>>>(Xf, out_back);
    write_forecast_kernel<<<cdiv((long long)Bc * Nn * HORc * Cc, 256), 256, 0, stream>>>(
        facc, level, tgf, out_fore);
}
